// LSTMEncoder_14946486190797
// MI455X (gfx1250) — matrix operand order repaired and checked by reading, not compiled, not run on hardware
//
#include <hip/hip_runtime.h>
#include <hip/hip_bf16.h>
#include <cstdint>

// Problem constants (match reference)
#define TSEQ  2048
#define BATCH 16
#define NF    768
#define NH    512
#define G4H   2048          // 4*H
#define H2    1024          // 2*H
#define MROWS (TSEQ*BATCH)  // 32768

typedef __bf16 bf16;
typedef __attribute__((ext_vector_type(16))) __bf16 v16bf;
typedef __attribute__((ext_vector_type(8)))  float  v8f;

union Frag { v16bf v; uint4 q[2]; };

// A-matrix fragment (16x32 bf16), source row-major (ld elements per row).
// Lanes 0-15 = M rows; VGPR0-3 hold K=0..7 (+8 for lanes 16-31), VGPR4-7 hold
// K=16..23 (+8). Two contiguous 16B loads per lane.
__device__ inline v16bf load_a_frag(const bf16* base, int ld, int lane) {
  int m = lane & 15, hh = lane >> 4;
  const bf16* p = base + (size_t)m * ld + 8 * hh;
  Frag f;
  f.q[0] = *(const uint4*)p;
  f.q[1] = *(const uint4*)(p + 16);
  return f.v;
}

// B-matrix fragment (32x16 bf16) where B[k][n] = W[n][k], W row-major (N x K).
// Lanes 0-15 = N cols hold K=0..7,16..23; lanes 16-31 hold K=8..15,24..31 -> one
// lane reads two 8-element runs from W's row n.
__device__ inline v16bf load_b_frag(const bf16* base, int ld, int lane) {
  int n = lane & 15, hh = lane >> 4;
  const bf16* p = base + (size_t)n * ld + 8 * hh;
  Frag f;
  f.q[0] = *(const uint4*)p;
  f.q[1] = *(const uint4*)(p + 16);
  return f.v;
}

__device__ inline float fast_sigmoid(float x) {
  return 1.f / (1.f + __expf(-x));
}
// Stable fast tanh: one v_exp_f32 + reciprocal, no library branches.
__device__ inline float fast_tanh(float x) {
  float t = __expf(-2.f * __builtin_fabsf(x));
  float r = (1.f - t) / (1.f + t);
  return __builtin_copysignf(r, x);
}

// ---------------------------------------------------------------------------
// fp32 -> bf16 packing
__global__ void pack_bf16(const float* __restrict__ src, bf16* __restrict__ dst, int n) {
  int i = blockIdx.x * blockDim.x + threadIdx.x;
  int stride = gridDim.x * blockDim.x;
  for (; i < n; i += stride) dst[i] = (bf16)src[i];
}

// ---------------------------------------------------------------------------
// D(M x N, f32) = A(M x K, bf16) @ W(N x K, bf16)^T + bias(N)
// Block: 256 threads = 8 waves, tile 32(M) x 256(N); wave tile 16 x 64.
// Double-buffered and manually unrolled x2 so the two fragment sets alternate
// with no register copies: loads for set B issue while set A feeds the WMMAs.
// Requires K % 64 == 0 (true for 768 / 1024).
__global__ __launch_bounds__(256) void wmma_gemm_bias(
    const bf16* __restrict__ A, int lda,
    const bf16* __restrict__ W, int ldw,
    const float* __restrict__ bias,
    float* __restrict__ D, int ldd, int K)
{
  int lane = threadIdx.x & 31;
  int wave = threadIdx.x >> 5;
  int mBase = blockIdx.y * 32 + (wave & 1) * 16;
  int nBase = blockIdx.x * 256 + (wave >> 1) * 64;

  const bf16* Ab = A + (size_t)mBase * lda;
  const bf16* Wb[4];
#pragma unroll
  for (int i = 0; i < 4; ++i) Wb[i] = W + (size_t)(nBase + 16 * i) * ldw;

  // Prologue: fragment 0 -> set 0
  v16bf a0 = load_a_frag(Ab, lda, lane);
  v16bf b0[4];
#pragma unroll
  for (int i = 0; i < 4; ++i) b0[i] = load_b_frag(Wb[i], ldw, lane);

  v8f acc[4] = {};
  int k = 32;
  // Steady state: pairs (f, f+1) for f = 1, 3, ..., F-3  (F = K/32, even)
  for (; k + 32 < K; k += 64) {
    v16bf a1 = load_a_frag(Ab + k, lda, lane);
    v16bf b1[4];
#pragma unroll
    for (int i = 0; i < 4; ++i) b1[i] = load_b_frag(Wb[i] + k, ldw, lane);
#pragma unroll
    for (int i = 0; i < 4; ++i)
      acc[i] = __builtin_amdgcn_wmma_f32_16x16x32_bf16(
          false, a0, false, b0[i], (short)0, acc[i], false, false);

    a0 = load_a_frag(Ab + k + 32, lda, lane);
#pragma unroll
    for (int i = 0; i < 4; ++i) b0[i] = load_b_frag(Wb[i] + k + 32, ldw, lane);
#pragma unroll
    for (int i = 0; i < 4; ++i)
      acc[i] = __builtin_amdgcn_wmma_f32_16x16x32_bf16(
          false, a1, false, b1[i], (short)0, acc[i], false, false);
  }
  // Epilogue: set 0 holds fragment F-2; load+compute fragment F-1.
  {
    v16bf a1 = load_a_frag(Ab + k, lda, lane);
    v16bf b1[4];
#pragma unroll
    for (int i = 0; i < 4; ++i) b1[i] = load_b_frag(Wb[i] + k, ldw, lane);
#pragma unroll
    for (int i = 0; i < 4; ++i)
      acc[i] = __builtin_amdgcn_wmma_f32_16x16x32_bf16(
          false, a0, false, b0[i], (short)0, acc[i], false, false);
#pragma unroll
    for (int i = 0; i < 4; ++i)
      acc[i] = __builtin_amdgcn_wmma_f32_16x16x32_bf16(
          false, a1, false, b1[i], (short)0, acc[i], false, false);
  }

  int hh = lane >> 4;
#pragma unroll
  for (int i = 0; i < 4; ++i) {
    int n = nBase + 16 * i + (lane & 15);
    float bv = bias[n];
#pragma unroll
    for (int r = 0; r < 8; ++r) {
      D[(size_t)(mBase + r + 8 * hh) * ldd + n] = acc[i][r] + bv;
    }
  }
}

// ---------------------------------------------------------------------------
// Persistent bidirectional LSTM scan for one layer. blockIdx.x = direction.
// 1024 threads = 32 waves; wave w owns hidden columns [16w, 16w+16) and
// computes all four gate tiles (i,f,g,o) for those columns, so the cell
// update stays entirely in the wave's registers. Only h (16x512 bf16 = 16KB)
// round-trips through LDS between steps. K-loop (16 fragments) is fully
// unrolled with paired double-buffering (no register copies).
__global__ __launch_bounds__(1024) void lstm_scan(
    const float* __restrict__ xpF, const float* __restrict__ xpB,
    const bf16* __restrict__ whhF, const bf16* __restrict__ whhB,
    bf16* __restrict__ xout)
{
  __shared__ bf16 s_h[BATCH * NH];  // 16 KB

  const int dir = blockIdx.x;
  const float* xp = dir ? xpB : xpF;
  const bf16* Whh = dir ? whhB : whhF;

  int tid  = threadIdx.x;
  int lane = tid & 31;
  int wave = tid >> 5;
  int hh   = lane >> 4;
  int jBase = wave * 16;            // hidden-column tile
  int jcol  = jBase + (lane & 15);  // this lane's hidden column

  const bf16* Wg[4];
#pragma unroll
  for (int g = 0; g < 4; ++g) Wg[g] = Whh + (size_t)(g * NH + jBase) * NH;

  for (int q = tid; q < BATCH * NH; q += 1024) s_h[q] = (bf16)0.0f;
  float c[8] = {0.f, 0.f, 0.f, 0.f, 0.f, 0.f, 0.f, 0.f};
  __syncthreads();

  for (int step = 0; step < TSEQ; ++step) {
    int t = dir ? (TSEQ - 1 - step) : step;

    // Prefetch next timestep's xp row (16x2048 f32 = 128KB; 128B per thread)
    // so it is L2/L1-resident when the elementwise phase needs it.
    if (step + 1 < TSEQ) {
      int tn = dir ? (t - 1) : (t + 1);
      const float* nxt = xp + (size_t)tn * BATCH * G4H + (size_t)tid * 32;
      __builtin_prefetch(nxt, 0, 3);
    }

    // gates(16 x 4H) = h @ Whh^T  (this wave: 4 gate tiles of same columns)
    v16bf a0 = load_a_frag(s_h, NH, lane);
    v16bf b0[4];
#pragma unroll
    for (int g = 0; g < 4; ++g) b0[g] = load_b_frag(Wg[g], NH, lane);

    v8f acc[4] = {};
#pragma unroll
    for (int k = 32; k + 32 < NH; k += 64) {
      v16bf a1 = load_a_frag(s_h + k, NH, lane);
      v16bf b1[4];
#pragma unroll
      for (int g = 0; g < 4; ++g) b1[g] = load_b_frag(Wg[g] + k, NH, lane);
#pragma unroll
      for (int g = 0; g < 4; ++g)
        acc[g] = __builtin_amdgcn_wmma_f32_16x16x32_bf16(
            false, a0, false, b0[g], (short)0, acc[g], false, false);

      a0 = load_a_frag(s_h + k + 32, NH, lane);
#pragma unroll
      for (int g = 0; g < 4; ++g) b0[g] = load_b_frag(Wg[g] + k + 32, NH, lane);
#pragma unroll
      for (int g = 0; g < 4; ++g)
        acc[g] = __builtin_amdgcn_wmma_f32_16x16x32_bf16(
            false, a1, false, b1[g], (short)0, acc[g], false, false);
    }
    {
      const int k = NH - 32;
      v16bf a1 = load_a_frag(s_h + k, NH, lane);
      v16bf b1[4];
#pragma unroll
      for (int g = 0; g < 4; ++g) b1[g] = load_b_frag(Wg[g] + k, NH, lane);
#pragma unroll
      for (int g = 0; g < 4; ++g)
        acc[g] = __builtin_amdgcn_wmma_f32_16x16x32_bf16(
            false, a0, false, b0[g], (short)0, acc[g], false, false);
#pragma unroll
      for (int g = 0; g < 4; ++g)
        acc[g] = __builtin_amdgcn_wmma_f32_16x16x32_bf16(
            false, a1, false, b1[g], (short)0, acc[g], false, false);
    }

    __syncthreads();  // all waves done reading s_h before it is rewritten

    const float* xprow = xp + (size_t)t * BATCH * G4H;
#pragma unroll
    for (int r = 0; r < 8; ++r) {
      int m = r + 8 * hh;  // batch row
      const float* xr = xprow + (size_t)m * G4H + jcol;
      float ig = acc[0][r] + xr[0];
      float fg = acc[1][r] + xr[NH];
      float gg = acc[2][r] + xr[2 * NH];
      float og = acc[3][r] + xr[3 * NH];
      float si = fast_sigmoid(ig);
      float sf = fast_sigmoid(fg);
      float so = fast_sigmoid(og);
      float tg = fast_tanh(gg);
      c[r] = sf * c[r] + si * tg;
      float h = so * fast_tanh(c[r]);
      s_h[m * NH + jcol] = (bf16)h;
      xout[((size_t)t * BATCH + m) * H2 + dir * NH + jcol] = (bf16)h;
    }
    __syncthreads();  // s_h fully updated before next step's GEMM
  }
}

// ---------------------------------------------------------------------------
// In-place LayerNorm over last dim (768) of d_out. One wave per row.
__global__ __launch_bounds__(256) void layernorm_inplace(
    float* __restrict__ y, const float* __restrict__ w, const float* __restrict__ b)
{
  int lane = threadIdx.x & 31;
  int row  = blockIdx.x * 8 + (threadIdx.x >> 5);
  float* p = y + (size_t)row * NF;

  float v[24];
  float s = 0.f, ss = 0.f;
#pragma unroll
  for (int i = 0; i < 24; ++i) {
    v[i] = p[lane + 32 * i];
    s += v[i];
    ss += v[i] * v[i];
  }
#pragma unroll
  for (int o = 16; o > 0; o >>= 1) {
    s  += __shfl_xor(s, o, 32);
    ss += __shfl_xor(ss, o, 32);
  }
  float mu  = s * (1.f / NF);
  float var = ss * (1.f / NF) - mu * mu;
  float rs  = rsqrtf(var + 1e-5f);
#pragma unroll
  for (int i = 0; i < 24; ++i) {
    int col = lane + 32 * i;
    p[col] = (v[i] - mu) * rs * w[col] + b[col];
  }
}

// ---------------------------------------------------------------------------
extern "C" void kernel_launch(void* const* d_in, const int* in_sizes, int n_in,
                              void* d_out, int out_size, void* d_ws, size_t ws_size,
                              hipStream_t stream)
{
  (void)in_sizes; (void)n_in; (void)out_size; (void)ws_size;

  // d_in layout: 0=inputs; 1+l*6+d*3+{0,1,2}={Wih,Whh,b}; 19=proj_w; 20=proj_b;
  // 21=ln_w; 22=ln_b
  const float* x_in  = (const float*)d_in[0];
  const float* projw = (const float*)d_in[19];
  const float* projb = (const float*)d_in[20];
  const float* lnw   = (const float*)d_in[21];
  const float* lnb   = (const float*)d_in[22];

  char* ws = (char*)d_ws;
  size_t off = 0;
  auto alloc = [&](size_t bytes) -> void* {
    void* p = ws + off;
    off = (off + bytes + 255) & ~(size_t)255;
    return p;
  };

  bf16* xbfA = (bf16*)alloc((size_t)MROWS * H2 * sizeof(bf16));
  bf16* xbfB = (bf16*)alloc((size_t)MROWS * H2 * sizeof(bf16));
  float* xpF = (float*)alloc((size_t)MROWS * G4H * sizeof(float));
  float* xpB = (float*)alloc((size_t)MROWS * G4H * sizeof(float));
  bf16* wih[3][2];
  bf16* whh[3][2];
  for (int l = 0; l < 3; ++l)
    for (int d = 0; d < 2; ++d) {
      int in_dim = l ? H2 : NF;
      wih[l][d] = (bf16*)alloc((size_t)G4H * in_dim * sizeof(bf16));
      whh[l][d] = (bf16*)alloc((size_t)G4H * NH * sizeof(bf16));
    }
  bf16* projwbf = (bf16*)alloc((size_t)NF * H2 * sizeof(bf16));

  auto packN = [&](const float* s, bf16* dst, size_t n) {
    int blocks = (int)((n + 2047) / 2048);  // ~8 elems/thread via grid-stride
    if (blocks > 8192) blocks = 8192;
    pack_bf16<<<dim3(blocks), dim3(256), 0, stream>>>(s, dst, (int)n);
  };

  // Pack activations (layer 0 input) + all weights to bf16.
  packN(x_in, xbfA, (size_t)MROWS * NF);
  for (int l = 0; l < 3; ++l)
    for (int d = 0; d < 2; ++d) {
      int in_dim = l ? H2 : NF;
      packN((const float*)d_in[1 + l * 6 + d * 3 + 0], wih[l][d], (size_t)G4H * in_dim);
      packN((const float*)d_in[1 + l * 6 + d * 3 + 1], whh[l][d], (size_t)G4H * NH);
    }
  packN(projw, projwbf, (size_t)NF * H2);

  // Layers
  const bf16* xin = xbfA;
  bf16* xout = xbfB;
  int inDim = NF;
  for (int l = 0; l < 3; ++l) {
    dim3 gg(G4H / 256, MROWS / 32);
    const float* bF = (const float*)d_in[1 + l * 6 + 0 * 3 + 2];
    const float* bB = (const float*)d_in[1 + l * 6 + 1 * 3 + 2];
    wmma_gemm_bias<<<gg, 256, 0, stream>>>(xin, inDim, wih[l][0], inDim, bF, xpF, G4H, inDim);
    wmma_gemm_bias<<<gg, 256, 0, stream>>>(xin, inDim, wih[l][1], inDim, bB, xpB, G4H, inDim);
    lstm_scan<<<dim3(2), dim3(1024), 0, stream>>>(xpF, xpB, whh[l][0], whh[l][1], xout);
    const bf16* tmp = xout;
    xout = (bf16*)xin;
    xin = tmp;
    inDim = H2;
  }

  // Projection GEMM into d_out, then in-place LayerNorm.
  dim3 gp(NF / 256, MROWS / 32);
  wmma_gemm_bias<<<gp, 256, 0, stream>>>(xin, H2, projwbf, H2, projb,
                                         (float*)d_out, NF, H2);
  layernorm_inplace<<<dim3(MROWS / 8), dim3(256), 0, stream>>>((float*)d_out, lnw, lnb);
}
